// AutoCorrelationAttention_34961033790159
// MI455X (gfx1250) — compile-verified, matrix operand order checked by reading, not run on hardware
//
#include <hip/hip_runtime.h>
#include <hip/hip_bf16.h>

typedef __attribute__((ext_vector_type(16))) _Float16 v16h;
typedef __attribute__((ext_vector_type(8)))  _Float16 v8h;
typedef __attribute__((ext_vector_type(8)))  float    v8f;
typedef __attribute__((ext_vector_type(4)))  float    v4f;

#define B_  8
#define L_  4096
#define E_  1024
#define H_  16
#define D_  64
#define BL_ (B_*L_)

#define NSTAGE 12          // log2(4096)
#define FFT_T  512         // threads per FFT workgroup (16 waves)
#define DSPLIT 8           // d-parallel blocks per (b,h) in forward FFT pass

// ---------------------------------------------------------------------------
// f32 -> f16 conversion (grid-stride)
// ---------------------------------------------------------------------------
__global__ void k_cvt_f16(const float* __restrict__ src, _Float16* __restrict__ dst, int n) {
  int i  = blockIdx.x * blockDim.x + threadIdx.x;
  int st = gridDim.x * blockDim.x;
  for (; i < n; i += st) dst[i] = (_Float16)src[i];
}

// ---------------------------------------------------------------------------
// QKV projection GEMM: Y[b,o,l] = sum_i x[b,l,i]*W[o,i] + bias[o]
// One wave computes a 64(l) x 64(o) tile with 4x4 WMMA accumulators.
// Output stored TRANSPOSED as [B][E][L] so FFT / attnV reads are contiguous.
// ---------------------------------------------------------------------------
__global__ __launch_bounds__(256) void k_gemm_qkv(
    const _Float16* __restrict__ xh,       // [BL][E] f16
    const _Float16* __restrict__ wh,       // [3][E][E] f16 (q,k,v)
    const float* __restrict__ bq, const float* __restrict__ bk, const float* __restrict__ bv,
    float* __restrict__ Qt, float* __restrict__ Kt, float* __restrict__ Vt)  // [B][E][L]
{
  const int lane = threadIdx.x & 31;
  const int gw   = blockIdx.x * 8 + (threadIdx.x >> 5);
  const int z    = gw >> 13;                 // 8192 tiles per matrix
  const int rem  = gw & 8191;
  const int tl   = rem >> 4;                 // 512 l-tiles of 64
  const int to   = rem & 15;                 // 16 o-tiles of 64
  const int n0   = tl * 64;                  // global token row
  const int o0   = to * 64;
  const int b    = n0 >> 12;                 // n0 / L
  const int l0   = n0 & (L_ - 1);

  const _Float16* W   = wh + (size_t)z * E_ * E_;
  const float* bias   = (z == 0) ? bq : (z == 1) ? bk : bv;
  float*       Yt     = (z == 0) ? Qt : (z == 1) ? Kt : Vt;

  const int kc  = (lane >> 4) << 3;          // lanes 16-31 take K+8 chunk
  const int col = lane & 15;

  v8f acc[4][4] = {};

  const _Float16* arow[4];
  const _Float16* brow[4];
  #pragma unroll
  for (int t = 0; t < 4; ++t) {
    arow[t] = xh + (size_t)(n0 + t * 16 + col) * E_;
    brow[t] = W  + (size_t)(o0 + t * 16 + col) * E_;
  }

  for (int k0 = 0; k0 < E_; k0 += 32) {
    const int kb = k0 + kc;
    v16h Af[4], Bf[4];
    #pragma unroll
    for (int t = 0; t < 4; ++t) {
      // documented 16-bit A/B fragment: lane holds K=[kb..kb+7] and [kb+16..kb+23]
      *(v8h*)&Af[t]         = *(const v8h*)(arow[t] + kb);
      *(((v8h*)&Af[t]) + 1) = *(const v8h*)(arow[t] + kb + 16);
      *(v8h*)&Bf[t]         = *(const v8h*)(brow[t] + kb);
      *(((v8h*)&Bf[t]) + 1) = *(const v8h*)(brow[t] + kb + 16);
    }
    if (k0 + 64 < E_) {                      // gfx1250 global_prefetch_b8 hint
      __builtin_prefetch(arow[0] + kb + 64, 0, 1);
      __builtin_prefetch(brow[0] + kb + 64, 0, 1);
    }
    #pragma unroll
    for (int mt = 0; mt < 4; ++mt)
      #pragma unroll
      for (int nt = 0; nt < 4; ++nt)
        acc[mt][nt] = __builtin_amdgcn_wmma_f32_16x16x32_f16(
            false, Af[mt], false, Bf[nt], (short)0, acc[mt][nt], false, false);
  }

  // Epilogue: bias add + transposed store. D-frag: lane=col N, VGPR v = row M
  // (lanes>=16 rows +8), so each lane writes 8 contiguous l's per tile.
  const int lrow = (lane >> 4) << 3;
  #pragma unroll
  for (int nt = 0; nt < 4; ++nt) {
    const int oc   = o0 + nt * 16 + col;
    const float bb = bias[oc];
    float* colp    = Yt + ((size_t)b * E_ + oc) * L_;
    #pragma unroll
    for (int mt = 0; mt < 4; ++mt) {
      const int lb = l0 + mt * 16 + lrow;
      v4f lo = { acc[mt][nt][0] + bb, acc[mt][nt][1] + bb, acc[mt][nt][2] + bb, acc[mt][nt][3] + bb };
      v4f hi = { acc[mt][nt][4] + bb, acc[mt][nt][5] + bb, acc[mt][nt][6] + bb, acc[mt][nt][7] + bb };
      *(v4f*)(colp + lb)     = lo;
      *(v4f*)(colp + lb + 4) = hi;
    }
  }
}

// ---------------------------------------------------------------------------
__global__ void k_zero(float* __restrict__ p, int n) {
  int i  = blockIdx.x * blockDim.x + threadIdx.x;
  int st = gridDim.x * blockDim.x;
  for (; i < n; i += st) p[i] = 0.f;
}

// ---------------------------------------------------------------------------
// shared radix-2 DIT FFT over LDS (input pre-scrambled in bit-reversed order)
// ---------------------------------------------------------------------------
__device__ __forceinline__ void fft4096(float2* buf, const float2* tw, int tid) {
  for (int s = 0; s < NSTAGE; ++s) {
    const int half = 1 << s;
    for (int u = tid; u < L_ / 2; u += FFT_T) {
      const int idx  = ((u >> s) << (s + 1)) | (u & (half - 1));
      const float2 w = tw[(u & (half - 1)) << (11 - s)];
      float2 a = buf[idx];
      float2 c = buf[idx + half];
      float2 t = make_float2(c.x * w.x - c.y * w.y, c.x * w.y + c.y * w.x);
      buf[idx]        = make_float2(a.x + t.x, a.y + t.y);
      buf[idx + half] = make_float2(a.x - t.x, a.y - t.y);
    }
    __syncthreads();
  }
}

// ---------------------------------------------------------------------------
// Forward pass: z_d = Q_d + i*K_d, Z = FFT(z), split Hermitian halves,
// accumulate S[f] += Qf*conj(Kf) over d (f32 atomics; irfft is linear).
// Grid: (B*H)*DSPLIT blocks, each handles D/DSPLIT = 8 sequences.
// ---------------------------------------------------------------------------
__global__ __launch_bounds__(FFT_T) void k_autocorr_fwd(
    const float* __restrict__ Qt, const float* __restrict__ Kt,
    float* __restrict__ S)                       // [B*H][2*L] interleaved re/im
{
  __shared__ float2 buf[L_];
  __shared__ float2 tw[L_ / 2];
  const int tid = threadIdx.x;
  const int bh  = blockIdx.x >> 3;
  const int dc  = blockIdx.x & (DSPLIT - 1);
  const int bb  = bh >> 4;
  const int hh  = bh & 15;

  for (int i = tid; i < L_ / 2; i += FFT_T) {
    float sn, cs;
    __sincosf(-6.283185307179586f * (float)i / (float)L_, &sn, &cs);
    tw[i] = make_float2(cs, sn);
  }
  __syncthreads();

  float2 pacc[8];
  #pragma unroll
  for (int j = 0; j < 8; ++j) pacc[j] = make_float2(0.f, 0.f);

  for (int dj = 0; dj < D_ / DSPLIT; ++dj) {
    const int d = dc * (D_ / DSPLIT) + dj;
    const size_t base = ((size_t)bb * E_ + hh * D_ + d) * L_;
    for (int i = tid; i < L_; i += FFT_T) {
      int j = __brev((unsigned)i) >> 20;       // 12-bit reversal
      buf[i] = make_float2(Qt[base + j], Kt[base + j]);
    }
    __syncthreads();
    fft4096(buf, tw, tid);
    #pragma unroll
    for (int j = 0; j < 8; ++j) {
      const int f = tid + j * FFT_T;
      float2 Zf = buf[f];
      float2 Zn = buf[(L_ - f) & (L_ - 1)];
      float2 Qf = make_float2(0.5f * (Zf.x + Zn.x),  0.5f * (Zf.y - Zn.y));
      float2 Kf = make_float2(0.5f * (Zf.y + Zn.y), -0.5f * (Zf.x - Zn.x));
      pacc[j].x += Qf.x * Kf.x + Qf.y * Kf.y;  // Qf * conj(Kf)
      pacc[j].y += Qf.y * Kf.x - Qf.x * Kf.y;
    }
    __syncthreads();
  }

  float* Sp = S + (size_t)bh * (2 * L_);
  #pragma unroll
  for (int j = 0; j < 8; ++j) {
    const int f = tid + j * FFT_T;
    atomicAdd(Sp + 2 * f,     pacc[j].x);
    atomicAdd(Sp + 2 * f + 1, pacc[j].y);
  }
}

// ---------------------------------------------------------------------------
// Inverse FFT (via FFT of conj) + fused softmax over L. One block per (b,h).
// ---------------------------------------------------------------------------
__global__ __launch_bounds__(FFT_T) void k_inv_softmax(
    const float* __restrict__ S, float* __restrict__ attn)   // attn [B*H][L]
{
  __shared__ float2 buf[L_];
  __shared__ float2 tw[L_ / 2];
  __shared__ float  red[FFT_T];
  const int tid = threadIdx.x;
  const int bh  = blockIdx.x;

  for (int i = tid; i < L_ / 2; i += FFT_T) {
    float sn, cs;
    __sincosf(-6.283185307179586f * (float)i / (float)L_, &sn, &cs);
    tw[i] = make_float2(cs, sn);
  }
  const float* Sp = S + (size_t)bh * (2 * L_);
  for (int i = tid; i < L_; i += FFT_T) {
    int j = __brev((unsigned)i) >> 20;
    buf[i] = make_float2(Sp[2 * j], -Sp[2 * j + 1]);   // conj -> inverse
  }
  __syncthreads();
  fft4096(buf, tw, tid);

  const float scale = 1.0f / ((float)L_ * (float)D_);  // irfft 1/L and mean over D
  float cv[8];
  float mx = -3.402823e38f;
  #pragma unroll
  for (int j = 0; j < 8; ++j) {
    cv[j] = buf[tid + j * FFT_T].x * scale;
    mx = fmaxf(mx, cv[j]);
  }
  red[tid] = mx; __syncthreads();
  for (int off = FFT_T / 2; off > 0; off >>= 1) {
    if (tid < off) red[tid] = fmaxf(red[tid], red[tid + off]);
    __syncthreads();
  }
  mx = red[0]; __syncthreads();

  float sum = 0.f;
  #pragma unroll
  for (int j = 0; j < 8; ++j) { cv[j] = __expf(cv[j] - mx); sum += cv[j]; }
  red[tid] = sum; __syncthreads();
  for (int off = FFT_T / 2; off > 0; off >>= 1) {
    if (tid < off) red[tid] += red[tid + off];
    __syncthreads();
  }
  const float inv = 1.0f / red[0];

  float* ap = attn + (size_t)bh * L_;
  #pragma unroll
  for (int j = 0; j < 8; ++j) ap[tid + j * FFT_T] = cv[j] * inv;
}

// ---------------------------------------------------------------------------
// out[b,h,d] = sum_l attn[b,h,l] * V[b,h,l,d]; store D-major: v2[b, d*H+h]
// One block per (b,o) channel; Vt contiguous along l.
// ---------------------------------------------------------------------------
__global__ __launch_bounds__(256) void k_attnv(
    const float* __restrict__ attn, const float* __restrict__ Vt, float* __restrict__ v2)
{
  __shared__ float red[256];
  const int tid = threadIdx.x;
  const int bo  = blockIdx.x;                // b*E + o
  const int b   = bo >> 10;
  const int o   = bo & (E_ - 1);
  const int h   = o >> 6;
  const int d   = o & 63;
  const float* vp = Vt + (size_t)bo * L_;
  const float* ap = attn + (size_t)(b * H_ + h) * L_;
  float s = 0.f;
  for (int l = tid; l < L_; l += 256) s += ap[l] * vp[l];
  red[tid] = s; __syncthreads();
  for (int off = 128; off > 0; off >>= 1) {
    if (tid < off) red[tid] += red[tid + off];
    __syncthreads();
  }
  if (tid == 0) v2[(size_t)b * E_ + (d * H_ + h)] = red[0];
}

// ---------------------------------------------------------------------------
// tiny output projection: proj[b,o] = sum_i v2[b,i]*Wo[o,i] + bo[o]
// ---------------------------------------------------------------------------
__global__ void k_outproj(const float* __restrict__ v2, const float* __restrict__ Wo,
                          const float* __restrict__ bo_, float* __restrict__ proj)
{
  int g = blockIdx.x * blockDim.x + threadIdx.x;
  if (g >= B_ * E_) return;
  const int b = g >> 10, o = g & (E_ - 1);
  const float* vr = v2 + (size_t)b * E_;
  const float* wr = Wo + (size_t)o * E_;
  float s = bo_[o];
  for (int i = 0; i < E_; ++i) s += vr[i] * wr[i];
  proj[g] = s;
}

// ---------------------------------------------------------------------------
// broadcast along L: out[b,l,:] = proj[b,:]  (128 MB write, float4 stores)
// ---------------------------------------------------------------------------
__global__ void k_bcast(const float* __restrict__ proj, float* __restrict__ out) {
  size_t i4 = (size_t)blockIdx.x * blockDim.x + threadIdx.x;
  const size_t n4 = (size_t)B_ * L_ * E_ / 4;
  const size_t st = (size_t)gridDim.x * blockDim.x;
  for (; i4 < n4; i4 += st) {
    const size_t idx = i4 * 4;
    const int e = (int)(idx & (E_ - 1));
    const int b = (int)(idx / ((size_t)L_ * E_));
    *(v4f*)(out + idx) = *(const v4f*)(proj + (size_t)b * E_ + e);
  }
}

// ---------------------------------------------------------------------------
extern "C" void kernel_launch(void* const* d_in, const int* in_sizes, int n_in,
                              void* d_out, int out_size, void* d_ws, size_t ws_size,
                              hipStream_t stream) {
  const float* x  = (const float*)d_in[0];
  const float* Wq = (const float*)d_in[1];
  const float* bq = (const float*)d_in[2];
  const float* Wk = (const float*)d_in[3];
  const float* bk = (const float*)d_in[4];
  const float* Wv = (const float*)d_in[5];
  const float* bv = (const float*)d_in[6];
  const float* Wo = (const float*)d_in[7];
  const float* bo = (const float*)d_in[8];
  float* out = (float*)d_out;

  char* ws = (char*)d_ws;
  size_t off = 0;
  auto alloc = [&](size_t bytes) -> void* {
    void* p = ws + off;
    off = (off + bytes + 255) & ~(size_t)255;
    return p;
  };
  _Float16* xh  = (_Float16*)alloc((size_t)BL_ * E_ * 2);          // 64 MB
  _Float16* wh  = (_Float16*)alloc((size_t)3 * E_ * E_ * 2);       // 6 MB
  float*    Qt  = (float*)alloc((size_t)B_ * E_ * L_ * 4);         // 128 MB
  float*    Kt  = (float*)alloc((size_t)B_ * E_ * L_ * 4);         // 128 MB
  float*    Vt  = (float*)alloc((size_t)B_ * E_ * L_ * 4);         // 128 MB
  float*    S   = (float*)alloc((size_t)B_ * H_ * 2 * L_ * 4);     // 4 MB
  float*    attn= (float*)alloc((size_t)B_ * H_ * L_ * 4);         // 2 MB
  float*    v2  = (float*)alloc((size_t)B_ * E_ * 4);
  float*    proj= (float*)alloc((size_t)B_ * E_ * 4);
  (void)ws_size; (void)in_sizes; (void)n_in; (void)out_size;

  k_cvt_f16<<<4096, 256, 0, stream>>>(x,  xh, BL_ * E_);
  k_cvt_f16<<<1024, 256, 0, stream>>>(Wq, wh,                         E_ * E_);
  k_cvt_f16<<<1024, 256, 0, stream>>>(Wk, wh + (size_t)E_ * E_,       E_ * E_);
  k_cvt_f16<<<1024, 256, 0, stream>>>(Wv, wh + (size_t)2 * E_ * E_,   E_ * E_);

  k_gemm_qkv<<<3072, 256, 0, stream>>>(xh, wh, bq, bk, bv, Qt, Kt, Vt);

  k_zero<<<512, 256, 0, stream>>>(S, B_ * H_ * 2 * L_);
  k_autocorr_fwd<<<B_ * H_ * DSPLIT, FFT_T, 0, stream>>>(Qt, Kt, S);
  k_inv_softmax<<<B_ * H_, FFT_T, 0, stream>>>(S, attn);

  k_attnv<<<B_ * E_, 256, 0, stream>>>(attn, Vt, v2);
  k_outproj<<<32, 256, 0, stream>>>(v2, Wo, bo, proj);
  k_bcast<<<8192, 256, 0, stream>>>(proj, out);
}
